// EGNN_VN_Encoder_point_cloud_24395414242132
// MI455X (gfx1250) — matrix operand order repaired and checked by reading, not compiled, not run on hardware
//
#include <hip/hip_runtime.h>
#include <math.h>

// ---------------------------------------------------------------------------
// EGNN + VN point-cloud encoder for MI455X (gfx1250, wave32, WMMA).
//   - Pairwise Gram matrices (kNN) via V_WMMA_F32_16X16X4_F32 on zero-padded
//     channel buffers (branchless K-loop, EXEC all-ones, fp32 keeps neighbor
//     ranking exact).
//   - All MLP GEMMs via V_WMMA_F32_16X16X32_F16: one wave computes a 16x64
//     tile (4 accumulators, A operand reused 4x), vectorized v8h/v16h loads.
//   - No atomics anywhere (deterministic): segment sums exploit the
//     row-sorted dense edge list (63 edges per atom) and per-atom gathers.
// ---------------------------------------------------------------------------

typedef _Float16 half_t;
typedef __attribute__((ext_vector_type(16))) _Float16 v16h;
typedef __attribute__((ext_vector_type(8)))  _Float16 v8h;
typedef __attribute__((ext_vector_type(8)))  float    v8f;
typedef __attribute__((ext_vector_type(2)))  float    v2f;

static constexpr int Bc   = 32;     // batch
static constexpr int Np   = 512;    // points per batch
static constexpr int NAc  = 64;     // atoms per batch
static constexpr int KNN  = 10;
static constexpr int Ec   = Bc * NAc * (NAc - 1);  // 129024 edges
static constexpr int NCc  = 64;
static constexpr int HDc  = 64;
static constexpr float SLOPEc = 0.2f;

enum { ACT_NONE = 0, ACT_SILU = 1, ACT_LRELU = 2 };

__device__ __forceinline__ float apply_act(float v, int act) {
  if (act == ACT_SILU)  return v / (1.f + expf(-v));
  if (act == ACT_LRELU) return (v >= 0.f) ? v : SLOPEc * v;
  return v;
}

// ---------------------------------------------------------------------------
__global__ void k_zero(float* __restrict__ p, int n) {
  int t = blockIdx.x * 256 + threadIdx.x;
  if (t < n) p[t] = 0.f;
}

// points (B*Np,3) -> x0 (B,4,Np)  (row 3 = zero pad, pre-zeroed)
__global__ void k_build_x0(const float* __restrict__ pts, float* __restrict__ x0) {
  int t = blockIdx.x * 256 + threadIdx.x;
  if (t >= Bc * Np) return;
  int bb = t / Np, i = t % Np;
  for (int c = 0; c < 3; ++c)
    x0[(size_t)bb * 4 * Np + c * Np + i] = pts[(size_t)t * 3 + c];
}

// xx[b,i] = sum_d xf[d,i]^2 over padded rows (pad rows are zero)
__global__ void k_xx(const float* __restrict__ xf, size_t ibs, int dpad,
                     float* __restrict__ xx) {
  int t = blockIdx.x * 256 + threadIdx.x;
  if (t >= Bc * Np) return;
  int bb = t / Np, i = t % Np;
  const float* xb = xf + (size_t)bb * ibs;
  float s = 0.f;
  for (int dd = 0; dd < dpad; ++dd) { float v = xb[dd * Np + i]; s += v * v; }
  xx[t] = s;
}

// neg_dist[b,i,j] = 2*inner - xx_i - xx_j via V_WMMA_F32_16X16X4_F32.
// dpad is a multiple of 4; pad rows are zero => branchless loads.
// grid (Np/16, Np/16, B), block 32 (one wave per 16x16 tile)
__global__ void k_inner_wmma(const float* __restrict__ xf, size_t ibs, int dpad,
                             const float* __restrict__ xx, float* __restrict__ dist) {
  int bb = blockIdx.z;
  int ib = blockIdx.x * 16, jb = blockIdx.y * 16;
  int l  = threadIdx.x;
  int lm = l & 15, hi = l >> 4;
  int mi = ib + lm, nj = jb + lm;
  const float* xb = xf + (size_t)bb * ibs;
  v8f acc = {};
#if defined(__HIP_DEVICE_COMPILE__) && __has_builtin(__builtin_amdgcn_wmma_f32_16x16x4_f32)
  const float* pa = xb + (size_t)(hi * 2) * Np + mi;
  const float* pb = xb + (size_t)(hi * 2) * Np + nj;
  for (int kk = 0; kk < dpad; kk += 4) {
    v2f av, bv;
    av[0] = pa[0]; av[1] = pa[Np];
    bv[0] = pb[0]; bv[1] = pb[Np];
    pa += 4 * Np; pb += 4 * Np;
    acc = __builtin_amdgcn_wmma_f32_16x16x4_f32(false, av, false, bv,
                                                (short)0, acc, false, false);
  }
#else
  for (int r = 0; r < 8; ++r) {
    int mrow = ib + r + hi * 8;
    float s = 0.f;
    for (int kk = 0; kk < dpad; ++kk) s += xb[kk * Np + mrow] * xb[kk * Np + nj];
    acc[r] = s;
  }
#endif
  const float* xxb = xx + bb * Np;
  float xxj = xxb[nj];
  float* db = dist + (size_t)bb * Np * Np;
  for (int r = 0; r < 8; ++r) {
    int mrow = ib + r + hi * 8;
    db[(size_t)mrow * Np + nj] = 2.f * acc[r] - xxb[mrow] - xxj;
  }
}

// top-k (k=10) per row, stable insertion (matches top_k tie order)
__global__ void k_topk(const float* __restrict__ dist, int* __restrict__ idxk) {
  int t = blockIdx.x * 256 + threadIdx.x;
  if (t >= Bc * Np) return;
  const float* row = dist + (size_t)t * Np;
  float bv[KNN]; int bi[KNN];
  for (int q = 0; q < KNN; ++q) { bv[q] = -3.4e38f; bi[q] = 0; }
  for (int j = 0; j < Np; ++j) {
    float v = row[j];
    if (v > bv[KNN - 1]) {
      int p = KNN - 1;
      while (p > 0 && bv[p - 1] < v) { bv[p] = bv[p - 1]; bi[p] = bi[p - 1]; --p; }
      bv[p] = v; bi[p] = j;
    }
  }
  for (int q = 0; q < KNN; ++q) idxk[t * KNN + q] = bi[q];
}

// fused get_graph_feature + vn_lin_leaky + mean over k.
// p = Wf_l @ nbr + (Wf_r - Wf_l) @ ctr   (same for d)
__global__ void k_vn_edge(const float* __restrict__ xin, size_t ibs, int cin,
                          const float* __restrict__ Wf, const float* __restrict__ Wd,
                          const int* __restrict__ idxk,
                          float* __restrict__ xout, size_t obs, int cout) {
  int t = blockIdx.x * 256 + threadIdx.x;
  if (t >= Bc * Np * cout) return;
  int co = t % cout; int rem = t / cout; int i = rem & (Np - 1); int bb = rem >> 9;
  const float* xb  = xin + (size_t)bb * ibs;
  const float* wfr = Wf + (size_t)co * 2 * cin;
  const float* wdr = Wd + (size_t)co * 2 * cin;
  float pc0 = 0, pc1 = 0, pc2 = 0, dc0 = 0, dc1 = 0, dc2 = 0;
  for (int ci = 0; ci < cin; ++ci) {
    float x0v = xb[(ci * 3 + 0) * Np + i];
    float x1v = xb[(ci * 3 + 1) * Np + i];
    float x2v = xb[(ci * 3 + 2) * Np + i];
    float fw = wfr[cin + ci] - wfr[ci];
    float dw = wdr[cin + ci] - wdr[ci];
    pc0 += fw * x0v; pc1 += fw * x1v; pc2 += fw * x2v;
    dc0 += dw * x0v; dc1 += dw * x1v; dc2 += dw * x2v;
  }
  const int* nb = idxk + (size_t)(bb * Np + i) * KNN;
  float o0 = 0, o1 = 0, o2 = 0;
  for (int e = 0; e < KNN; ++e) {
    int j = nb[e];
    float p0 = pc0, p1 = pc1, p2 = pc2, d0 = dc0, d1 = dc1, d2 = dc2;
    for (int ci = 0; ci < cin; ++ci) {
      float xj0 = xb[(ci * 3 + 0) * Np + j];
      float xj1 = xb[(ci * 3 + 1) * Np + j];
      float xj2 = xb[(ci * 3 + 2) * Np + j];
      float fw = wfr[ci], dw = wdr[ci];
      p0 += fw * xj0; p1 += fw * xj1; p2 += fw * xj2;
      d0 += dw * xj0; d1 += dw * xj1; d2 += dw * xj2;
    }
    float dot = p0 * d0 + p1 * d1 + p2 * d2;
    float dsq = d0 * d0 + d1 * d1 + d2 * d2;
    float coef = (dot >= 0.f) ? 0.f : ((1.f - SLOPEc) * dot / (dsq + 1e-6f));
    o0 += p0 - coef * d0; o1 += p1 - coef * d1; o2 += p2 - coef * d2;
  }
  const float sc = 1.f / (float)KNN;
  float* ob = xout + (size_t)bb * obs + (size_t)co * 3 * Np + i;
  ob[0] = o0 * sc; ob[Np] = o1 * sc; ob[2 * Np] = o2 * sc;
}

// conv5: vn_lin_leaky over the virtual concat [x1|x2|x3|x4] (169 ch)
__global__ void k_vn_conv5(const float* __restrict__ x1, const float* __restrict__ x2,
                           const float* __restrict__ x3, const float* __restrict__ x4,
                           const float* __restrict__ Wf, const float* __restrict__ Wd,
                           float* __restrict__ x5) {
  int t = blockIdx.x * 256 + threadIdx.x;
  if (t >= Bc * Np * NCc) return;
  int co = t % NCc; int rem = t / NCc; int i = rem & (Np - 1); int bb = rem >> 9;
  float p0 = 0, p1 = 0, p2 = 0, d0 = 0, d1 = 0, d2 = 0;
  const float* srcs[4] = { x1 + (size_t)bb * 64 * Np, x2 + (size_t)bb * 64 * Np,
                           x3 + (size_t)bb * 128 * Np, x4 + (size_t)bb * 255 * Np };
  const int cnt[4] = {21, 21, 42, 85};
  int cbase = 0;
  for (int s = 0; s < 4; ++s) {
    const float* xb = srcs[s];
    for (int ci = 0; ci < cnt[s]; ++ci) {
      float x0v = xb[(ci * 3 + 0) * Np + i];
      float x1v = xb[(ci * 3 + 1) * Np + i];
      float x2v = xb[(ci * 3 + 2) * Np + i];
      float fw = Wf[co * 169 + cbase + ci], dw = Wd[cbase + ci];
      p0 += fw * x0v; p1 += fw * x1v; p2 += fw * x2v;
      d0 += dw * x0v; d1 += dw * x1v; d2 += dw * x2v;
    }
    cbase += cnt[s];
  }
  float dot = p0 * d0 + p1 * d1 + p2 * d2;
  float dsq = d0 * d0 + d1 * d1 + d2 * d2;
  float coef = (dot >= 0.f) ? 0.f : ((1.f - SLOPEc) * dot / (dsq + 1e-6f));
  size_t base = ((size_t)(bb * NCc + co) * 3) * Np + i;
  x5[base]          = p0 - coef * d0;
  x5[base +     Np] = p1 - coef * d1;
  x5[base + 2 * Np] = p2 - coef * d2;
}

// deterministic scatter-mean points->atoms (gather form)
__global__ void k_scatter_gather(const float* __restrict__ x5, const int* __restrict__ pai,
                                 float* __restrict__ xatom) {
  int t = blockIdx.x * 256 + threadIdx.x;
  if (t >= Bc * NAc * NCc) return;
  int co = t % NCc; int rem = t / NCc; int a = rem & (NAc - 1); int bb = rem >> 6;
  const float* xb = x5 + ((size_t)(bb * NCc + co) * 3) * Np;
  const int* pb = pai + bb * Np;
  float s0 = 0, s1 = 0, s2 = 0; int cnt = 0;
  for (int i = 0; i < Np; ++i) {
    if (pb[i] == a) { ++cnt; s0 += xb[i]; s1 += xb[Np + i]; s2 += xb[2 * Np + i]; }
  }
  float inv = 1.f / fmaxf((float)cnt, 1.f);
  int base = ((bb * NAc + a) * NCc + co) * 3;
  xatom[base] = s0 * inv; xatom[base + 1] = s1 * inv; xatom[base + 2] = s2 * inv;
}

// Z_equivariant = sum over atoms; also kept for the VN-std mean channel
__global__ void k_zeq(const float* __restrict__ xatom, float* __restrict__ zeq,
                      float* __restrict__ dout) {
  int t = blockIdx.x * 256 + threadIdx.x;
  if (t >= Bc * NCc) return;
  int bb = t / NCc, co = t % NCc;
  for (int c = 0; c < 3; ++c) {
    float s = 0.f;
    for (int a = 0; a < NAc; ++a) s += xatom[((bb * NAc + a) * NCc + co) * 3 + c];
    zeq[(bb * NCc + co) * 3 + c]  = s;
    dout[(bb * NCc + co) * 3 + c] = s;
  }
}

// build xs (b,128,3,na) = concat(x_atom, broadcast sum)
__global__ void k_xs(const float* __restrict__ xatom, const float* __restrict__ zeq,
                     float* __restrict__ xs) {
  int t = blockIdx.x * 256 + threadIdx.x;
  if (t >= Bc * 128 * 3 * NAc) return;
  int m = t & (NAc - 1); int comp = (t >> 6) % 3; int c = (t / 192) % 128; int bb = t / (192 * 128);
  float v = (c < NCc) ? xatom[((bb * NAc + m) * NCc + c) * 3 + comp]
                      : zeq[(bb * NCc + (c - NCc)) * 3 + comp];
  xs[((size_t)(bb * 128 + c) * 3 + comp) * NAc + m] = v;
}

// vn_lin_leaky over atoms: in (b,cin,3,na) -> out (b,cout,3,na)
__global__ void k_vnll_atom(const float* __restrict__ xin, int cin,
                            const float* __restrict__ Wf, const float* __restrict__ Wd,
                            float* __restrict__ xout, int cout) {
  int t = blockIdx.x * 256 + threadIdx.x;
  if (t >= Bc * NAc * cout) return;
  int co = t % cout; int rem = t / cout; int m = rem & (NAc - 1); int bb = rem >> 6;
  const float* xb = xin + (size_t)bb * cin * 3 * NAc;
  float p0 = 0, p1 = 0, p2 = 0, d0 = 0, d1 = 0, d2 = 0;
  for (int ci = 0; ci < cin; ++ci) {
    float x0v = xb[(ci * 3 + 0) * NAc + m];
    float x1v = xb[(ci * 3 + 1) * NAc + m];
    float x2v = xb[(ci * 3 + 2) * NAc + m];
    float fw = Wf[co * cin + ci], dw = Wd[co * cin + ci];
    p0 += fw * x0v; p1 += fw * x1v; p2 += fw * x2v;
    d0 += dw * x0v; d1 += dw * x1v; d2 += dw * x2v;
  }
  float dot = p0 * d0 + p1 * d1 + p2 * d2;
  float dsq = d0 * d0 + d1 * d1 + d2 * d2;
  float coef = (dot >= 0.f) ? 0.f : ((1.f - SLOPEc) * dot / (dsq + 1e-6f));
  size_t base = ((size_t)(bb * cout + co) * 3) * NAc + m;
  xout[base]           = p0 - coef * d0;
  xout[base + NAc]     = p1 - coef * d1;
  xout[base + 2 * NAc] = p2 - coef * d2;
}

// z0 (b,3,3,na) = lin(3,32) @ z2 (b,32,3,na)
__global__ void k_z0(const float* __restrict__ z2, const float* __restrict__ lin,
                     float* __restrict__ z0b) {
  int t = blockIdx.x * 256 + threadIdx.x;
  if (t >= Bc * NAc * 3) return;
  int j = t % 3; int rem = t / 3; int m = rem & (NAc - 1); int bb = rem >> 6;
  for (int kc = 0; kc < 3; ++kc) {
    float s = 0.f;
    for (int ci = 0; ci < 32; ++ci)
      s += lin[j * 32 + ci] * z2[((size_t)(bb * 32 + ci) * 3 + kc) * NAc + m];
    z0b[((size_t)(bb * 3 + j) * 3 + kc) * NAc + m] = s;
  }
}

// x_inv = einsum('bijm,bjkm->bikm', xs, z0)
__global__ void k_xinv(const float* __restrict__ xs, const float* __restrict__ z0b,
                       float* __restrict__ xinv) {
  int t = blockIdx.x * 256 + threadIdx.x;
  if (t >= Bc * NAc * 128) return;
  int i = t % 128; int rem = t / 128; int m = rem & (NAc - 1); int bb = rem >> 6;
  for (int kc = 0; kc < 3; ++kc) {
    float s = 0.f;
    for (int j = 0; j < 3; ++j)
      s += xs[((size_t)(bb * 128 + i) * 3 + j) * NAc + m] *
           z0b[((size_t)(bb * 3 + j) * 3 + kc) * NAc + m];
    xinv[((size_t)(bb * 128 + i) * 3 + kc) * NAc + m] = s;
  }
}

// ---------------------------------------------------------------------------
// EGNN helpers
__global__ void k_radial(const int* __restrict__ ei, const float* __restrict__ pos,
                         float* __restrict__ rad) {
  int g = blockIdx.x * 256 + threadIdx.x;
  if (g >= Ec) return;
  int r = ei[g], c = ei[Ec + g];
  float s = 0.f;
  for (int q = 0; q < 3; ++q) { float d = pos[r * 3 + q] - pos[c * 3 + q]; s += d * d; }
  rad[g] = s;
}

// layer-0 edge features: [h_row(109) | h_col(109) | r2(1) | edge_attr(5)] -> f16 (E,224)
__global__ void k_ein0(const float* __restrict__ h, const int* __restrict__ ei,
                       const float* __restrict__ rad, const float* __restrict__ ea,
                       half_t* __restrict__ ein) {
  int t = blockIdx.x * 256 + threadIdx.x;
  if (t >= Ec * 224) return;
  int q = t % 224; int g = t / 224;
  float v;
  if (q < 109)      v = h[ei[g] * 109 + q];
  else if (q < 218) v = h[ei[Ec + g] * 109 + (q - 109)];
  else if (q == 218) v = rad[g];
  else              v = ea[(size_t)g * 5 + (q - 219)];
  ein[t] = (half_t)v;
}

// layer-1/2 edge features: [hh_row(64) | hh_col(64) | r2(1) | m(64) | pad] -> f16 (E,224)
__global__ void k_ein12(const float* __restrict__ hh, const int* __restrict__ ei,
                        const float* __restrict__ rad, const half_t* __restrict__ m16,
                        half_t* __restrict__ ein) {
  int t = blockIdx.x * 256 + threadIdx.x;
  if (t >= Ec * 224) return;
  int q = t % 224; int g = t / 224;
  float v;
  if (q < 64)        v = hh[ei[g] * 64 + q];
  else if (q < 128)  v = hh[ei[Ec + g] * 64 + (q - 64)];
  else if (q == 128) v = rad[g];
  else if (q < 193)  v = (float)m16[(size_t)g * 64 + (q - 129)];
  else               v = 0.f;
  ein[t] = (half_t)v;
}

// segment-sum over the 63 contiguous edges of each atom (row-sorted edge list)
__global__ void k_agg(const half_t* __restrict__ m16, float* __restrict__ agg) {
  int t = blockIdx.x * 256 + threadIdx.x;
  if (t >= Bc * NAc * HDc) return;
  int q = t % HDc; int a = t / HDc;
  float s = 0.f;
  for (int e = 0; e < NAc - 1; ++e) s += (float)m16[((size_t)a * 63 + e) * HDc + q];
  agg[t] = s;
}

// node features: [h|agg] padded -> f16 (2048, Kpad)
__global__ void k_nin(const float* __restrict__ A, int wA, const float* __restrict__ agg,
                      int Kpad, half_t* __restrict__ nin) {
  int t = blockIdx.x * 256 + threadIdx.x;
  if (t >= Bc * NAc * Kpad) return;
  int q = t % Kpad; int r = t / Kpad;
  float v = (q < wA) ? A[(size_t)r * wA + q]
                     : ((q < wA + HDc) ? agg[r * HDc + (q - wA)] : 0.f);
  nin[t] = (half_t)v;
}

// feat (b,na,448) f16: [x_inv flattened (384) | hh (64)]
__global__ void k_feat(const float* __restrict__ xinv, const float* __restrict__ hh,
                       half_t* __restrict__ feat) {
  int t = blockIdx.x * 256 + threadIdx.x;
  if (t >= Bc * NAc * 448) return;
  int f = t % 448; int r = t / 448; int m = r & (NAc - 1); int bb = r >> 6;
  float v;
  if (f < 384) {
    int i = f / 3, kc = f % 3;
    v = xinv[((size_t)(bb * 128 + i) * 3 + kc) * NAc + m];
  } else {
    v = hh[r * HDc + (f - 384)];
  }
  feat[t] = (half_t)v;
}

__global__ void k_sumatoms(const float* __restrict__ pw, float* __restrict__ zi0) {
  int t = blockIdx.x * 256 + threadIdx.x;
  if (t >= Bc * HDc) return;
  int bb = t / HDc, q = t % HDc;
  float s = 0.f;
  for (int m = 0; m < NAc; ++m) s += pw[(size_t)(bb * NAc + m) * HDc + q];
  zi0[t] = s;
}

__global__ void k_dense_small(const float* __restrict__ in, const float* __restrict__ W,
                              const float* __restrict__ bias, int M, int Kd, int Nd,
                              int act, float* __restrict__ out) {
  int t = blockIdx.x * 256 + threadIdx.x;
  if (t >= M * Nd) return;
  int r = t / Nd, q = t % Nd;
  float s = bias[q];
  for (int k = 0; k < Kd; ++k) s += in[r * Kd + k] * W[q * Kd + k];
  out[t] = apply_act(s, act);
}

// f32 (N,K) -> f16 (N,Kpad) zero-padded
__global__ void k_cvt_pad(const float* __restrict__ src, half_t* __restrict__ dst,
                          int K, int Kpad, int total) {
  int t = blockIdx.x * 256 + threadIdx.x;
  if (t >= total) return;
  int q = t % Kpad, r = t / Kpad;
  dst[t] = (half_t)((q < K) ? src[r * K + q] : 0.f);
}

// ---------------------------------------------------------------------------
// f16 WMMA GEMM: C[M,N] = act(A[M,K] * W[N,K]^T + bias) (+resid)
// One wave computes a 16x64 tile (4 accumulators, A reused 4x).
// grid (M/16, N/64), block 32. K, lda, ldw multiples of 16; K mult of 32.
__global__ void k_gemm_f16(const half_t* __restrict__ A, int lda,
                           const half_t* __restrict__ W, int ldw,
                           const float* __restrict__ bias, int K, int act,
                           float* __restrict__ outF, half_t* __restrict__ outH,
                           const float* __restrict__ resid, int ldo) {
  int l = threadIdx.x;
  int Mb = blockIdx.x * 16, Nb = blockIdx.y * 64;
  int lm = l & 15, hi = l >> 4;
  v8f acc[4] = {};
#if defined(__HIP_DEVICE_COMPILE__) && __has_builtin(__builtin_amdgcn_wmma_f32_16x16x32_f16)
  const half_t* Ar = A + (size_t)(Mb + lm) * lda + hi * 16;
  const half_t* Wr = W + (size_t)(Nb + lm) * ldw + hi * 16;
  const size_t wstep = (size_t)16 * ldw;
  for (int kk = 0; kk < K; kk += 32) {
    v8h alo = *(const v8h*)(Ar + kk);
    v8h ahi = *(const v8h*)(Ar + kk + 8);
    v16h av = __builtin_shufflevector(alo, ahi,
                                      0, 1, 2, 3, 4, 5, 6, 7,
                                      8, 9, 10, 11, 12, 13, 14, 15);
#pragma unroll
    for (int nt = 0; nt < 4; ++nt) {
      v16h bv = *(const v16h*)(Wr + nt * wstep + kk);
      acc[nt] = __builtin_amdgcn_wmma_f32_16x16x32_f16(false, av, false, bv,
                                                       (short)0, acc[nt], false, false);
    }
  }
#else
  for (int nt = 0; nt < 4; ++nt)
    for (int r = 0; r < 8; ++r) {
      int row = Mb + r + hi * 8, col = Nb + nt * 16 + lm;
      float s = 0.f;
      for (int k = 0; k < K; ++k)
        s += (float)A[(size_t)row * lda + k] * (float)W[(size_t)col * ldw + k];
      acc[nt][r] = s;
    }
#endif
#pragma unroll
  for (int nt = 0; nt < 4; ++nt) {
    int col = Nb + nt * 16 + lm;
    float bc = bias ? bias[col] : 0.f;
#pragma unroll
    for (int r = 0; r < 8; ++r) {
      int row = Mb + r + hi * 8;
      float v = apply_act(acc[nt][r] + bc, act);
      if (resid) v += resid[(size_t)row * ldo + col];
      if (outF) outF[(size_t)row * ldo + col] = v;
      if (outH) outH[(size_t)row * ldo + col] = (half_t)v;
    }
  }
}

// ---------------------------------------------------------------------------
extern "C" void kernel_launch(void* const* d_in, const int* in_sizes, int n_in,
                              void* d_out, int out_size, void* d_ws, size_t ws_size,
                              hipStream_t stream) {
  (void)in_sizes; (void)n_in; (void)out_size; (void)ws_size;
  auto F = [&](int i) { return (const float*)d_in[i]; };
  auto I = [&](int i) { return (const int*)d_in[i]; };

  const float* Hin = F(0);
  const int*   EI  = I(1);
  const float* POS = F(2);
  const float* EA  = F(3);
  const float* PTS = F(5);
  const int*   PAI = I(6);
  float* dout = (float*)d_out;

  // ---- workspace bump allocator (256B aligned) ----
  size_t off = 0;
  auto alloc = [&](size_t bytes) {
    void* p = (char*)d_ws + off;
    off += (bytes + 255) & ~(size_t)255;
    return p;
  };
  // VN feature buffers, channel-padded to a multiple of 4 rows (zeroed pads).
  // X0P..X3 are allocated contiguously so one k_zero covers all pad rows.
  float* X0P  = (float*)alloc((size_t)Bc * 4 * Np * 4);     // 3 used + 1 pad
  float* X1   = (float*)alloc((size_t)Bc * 64 * Np * 4);    // 63 used + 1 pad
  float* X2   = (float*)alloc((size_t)Bc * 64 * Np * 4);    // 63 used + 1 pad
  float* X3   = (float*)alloc((size_t)Bc * 128 * Np * 4);   // 126 used + 2 pad
  float* X4   = (float*)alloc((size_t)Bc * 255 * Np * 4);   // 255 (no dist needed)
  const int ZERO_N = Bc * (4 + 64 + 64 + 128) * Np;         // X0P..X3 contiguous
  float* XX    = (float*)alloc((size_t)Bc * Np * 4);
  float* DIST  = (float*)alloc((size_t)Bc * Np * Np * 4);
  int*   IDXK  = (int*)alloc((size_t)Bc * Np * KNN * 4);
  float* X5    = (float*)alloc((size_t)Bc * NCc * 3 * Np * 4);
  float* XATOM = (float*)alloc((size_t)Bc * NAc * NCc * 3 * 4);
  float* ZEQ   = (float*)alloc((size_t)Bc * NCc * 3 * 4);
  float* XS    = (float*)alloc((size_t)Bc * 128 * 3 * NAc * 4);
  float* Z1    = (float*)alloc((size_t)Bc * 64 * 3 * NAc * 4);
  float* Z2    = (float*)alloc((size_t)Bc * 32 * 3 * NAc * 4);
  float* Z0B   = (float*)alloc((size_t)Bc * 3 * 3 * NAc * 4);
  float* XINV  = (float*)alloc((size_t)Bc * 128 * 3 * NAc * 4);
  float* RAD   = (float*)alloc((size_t)Ec * 4);
  half_t* EIN16  = (half_t*)alloc((size_t)Ec * 224 * 2);
  half_t* MID16  = (half_t*)alloc((size_t)Ec * HDc * 2);
  half_t* M16    = (half_t*)alloc((size_t)Ec * HDc * 2);
  float*  AGG    = (float*)alloc((size_t)Bc * NAc * HDc * 4);
  half_t* NIN16  = (half_t*)alloc((size_t)Bc * NAc * 192 * 2);
  half_t* NMID16 = (half_t*)alloc((size_t)Bc * NAc * HDc * 2);
  float*  HH     = (float*)alloc((size_t)Bc * NAc * HDc * 4);
  half_t* FEAT16 = (half_t*)alloc((size_t)Bc * NAc * 448 * 2);
  half_t* PIH1   = (half_t*)alloc((size_t)Bc * NAc * 128 * 2);
  half_t* PIH2   = (half_t*)alloc((size_t)Bc * NAc * 64 * 2);
  float*  PW     = (float*)alloc((size_t)Bc * NAc * 64 * 4);
  float*  ZI0    = (float*)alloc((size_t)Bc * 64 * 4);
  float*  ZI1    = (float*)alloc((size_t)Bc * 64 * 4);
  float*  ZI2    = (float*)alloc((size_t)Bc * 64 * 4);
  half_t* EW1h[3], *EW2h[3], *NW1h[3], *NW2h[3];
  for (int l = 0; l < 3; ++l) {
    EW1h[l] = (half_t*)alloc((size_t)64 * 224 * 2);
    EW2h[l] = (half_t*)alloc((size_t)64 * 64 * 2);
    NW1h[l] = (half_t*)alloc((size_t)64 * 192 * 2);
    NW2h[l] = (half_t*)alloc((size_t)64 * 64 * 2);
  }
  half_t* PIW1h = (half_t*)alloc((size_t)128 * 448 * 2);
  half_t* PIW2h = (half_t*)alloc((size_t)64 * 128 * 2);
  half_t* PIW3h = (half_t*)alloc((size_t)64 * 64 * 2);

  auto g1 = [](int n) { return dim3((unsigned)((n + 255) / 256)); };
  auto cvt = [&](const float* src, half_t* dst, int K, int Kpad, int Nrows) {
    int tot = Nrows * Kpad;
    k_cvt_pad<<<g1(tot), 256, 0, stream>>>(src, dst, K, Kpad, tot);
  };
  auto gemm = [&](const half_t* A, int lda, const half_t* W, int ldw, const float* bias,
                  int M, int Nn, int K, int act, float* outF, half_t* outH,
                  const float* resid) {
    dim3 g((unsigned)(M / 16), (unsigned)(Nn / 64), 1);
    k_gemm_f16<<<g, 32, 0, stream>>>(A, lda, W, ldw, bias, K, act, outF, outH, resid, Nn);
  };

  // ---- f16 weight prep ----
  const int Ke[3] = {224, 193, 193}, Kn[3] = {173, 128, 128}, KnPad[3] = {192, 128, 128};
  for (int l = 0; l < 3; ++l) {
    int base = 34 + 8 * l;
    cvt(F(base + 0), EW1h[l], Ke[l], 224, 64);
    cvt(F(base + 2), EW2h[l], 64, 64, 64);
    cvt(F(base + 4), NW1h[l], Kn[l], KnPad[l], 64);
    cvt(F(base + 6), NW2h[l], 64, 64, 64);
  }
  cvt(F(22), PIW1h, 448, 448, 128);
  cvt(F(24), PIW2h, 128, 128, 64);
  cvt(F(26), PIW3h, 64, 64, 64);

  // ---- VN-DGCNN point branch ----
  k_zero<<<g1(ZERO_N), 256, 0, stream>>>(X0P, ZERO_N);  // zero pads of X0P..X3
  k_build_x0<<<g1(Bc * Np), 256, 0, stream>>>(PTS, X0P);
  struct Lyr { const float* in; size_t ibs; int cin, dpad; int wf, wd;
               float* out; size_t obs; int cout; };
  Lyr lyr[4] = {
    {X0P, (size_t)4 * Np,   1,  4,   7,  8, X1, (size_t)64 * Np,  21},
    {X1,  (size_t)64 * Np,  21, 64,  9, 10, X2, (size_t)64 * Np,  21},
    {X2,  (size_t)64 * Np,  21, 64, 11, 12, X3, (size_t)128 * Np, 42},
    {X3,  (size_t)128 * Np, 42, 128, 13, 14, X4, (size_t)255 * Np, 85},
  };
  for (int li = 0; li < 4; ++li) {
    k_xx<<<g1(Bc * Np), 256, 0, stream>>>(lyr[li].in, lyr[li].ibs, lyr[li].dpad, XX);
    k_inner_wmma<<<dim3(Np / 16, Np / 16, Bc), 32, 0, stream>>>(
        lyr[li].in, lyr[li].ibs, lyr[li].dpad, XX, DIST);
    k_topk<<<g1(Bc * Np), 256, 0, stream>>>(DIST, IDXK);
    k_vn_edge<<<g1(Bc * Np * lyr[li].cout), 256, 0, stream>>>(
        lyr[li].in, lyr[li].ibs, lyr[li].cin, F(lyr[li].wf), F(lyr[li].wd), IDXK,
        lyr[li].out, lyr[li].obs, lyr[li].cout);
  }
  k_vn_conv5<<<g1(Bc * Np * NCc), 256, 0, stream>>>(X1, X2, X3, X4, F(15), F(16), X5);
  k_scatter_gather<<<g1(Bc * NAc * NCc), 256, 0, stream>>>(X5, PAI, XATOM);
  k_zeq<<<g1(Bc * NCc), 256, 0, stream>>>(XATOM, ZEQ, dout);

  // ---- EGNN (3 layers) ----
  k_radial<<<g1(Ec), 256, 0, stream>>>(EI, POS, RAD);
  // layer 0
  k_ein0<<<g1(Ec * 224), 256, 0, stream>>>(Hin, EI, RAD, EA, EIN16);
  gemm(EIN16, 224, EW1h[0], 224, F(35), Ec, 64, 224, ACT_SILU, nullptr, MID16, nullptr);
  gemm(MID16, 64, EW2h[0], 64, F(37), Ec, 64, 64, ACT_SILU, nullptr, M16, nullptr);
  k_agg<<<g1(Bc * NAc * HDc), 256, 0, stream>>>(M16, AGG);
  k_nin<<<g1(Bc * NAc * 192), 256, 0, stream>>>(Hin, 109, AGG, 192, NIN16);
  gemm(NIN16, 192, NW1h[0], 192, F(39), Bc * NAc, 64, 192, ACT_SILU, nullptr, NMID16, nullptr);
  gemm(NMID16, 64, NW2h[0], 64, F(41), Bc * NAc, 64, 64, ACT_NONE, HH, nullptr, nullptr);
  // layers 1..2 (residual)
  for (int l = 1; l < 3; ++l) {
    int base = 34 + 8 * l;
    k_ein12<<<g1(Ec * 224), 256, 0, stream>>>(HH, EI, RAD, M16, EIN16);
    gemm(EIN16, 224, EW1h[l], 224, F(base + 1), Ec, 64, 224, ACT_SILU, nullptr, MID16, nullptr);
    gemm(MID16, 64, EW2h[l], 64, F(base + 3), Ec, 64, 64, ACT_SILU, nullptr, M16, nullptr);
    k_agg<<<g1(Bc * NAc * HDc), 256, 0, stream>>>(M16, AGG);
    k_nin<<<g1(Bc * NAc * 128), 256, 0, stream>>>(HH, 64, AGG, 128, NIN16);
    gemm(NIN16, 128, NW1h[l], 128, F(base + 5), Bc * NAc, 64, 128, ACT_SILU, nullptr, NMID16, nullptr);
    gemm(NMID16, 64, NW2h[l], 64, F(base + 7), Bc * NAc, 64, 64, ACT_NONE, HH, nullptr, HH);
  }

  // ---- VN std (invariant) feature ----
  k_xs<<<g1(Bc * 128 * 3 * NAc), 256, 0, stream>>>(XATOM, ZEQ, XS);
  k_vnll_atom<<<g1(Bc * NAc * 64), 256, 0, stream>>>(XS, 128, F(17), F(18), Z1, 64);
  k_vnll_atom<<<g1(Bc * NAc * 32), 256, 0, stream>>>(Z1, 64, F(19), F(20), Z2, 32);
  k_z0<<<g1(Bc * NAc * 3), 256, 0, stream>>>(Z2, F(21), Z0B);
  k_xinv<<<g1(Bc * NAc * 128), 256, 0, stream>>>(XS, Z0B, XINV);

  // ---- point-invariant MLP + pooled head ----
  k_feat<<<g1(Bc * NAc * 448), 256, 0, stream>>>(XINV, HH, FEAT16);
  gemm(FEAT16, 448, PIW1h, 448, F(23), Bc * NAc, 128, 448, ACT_LRELU, nullptr, PIH1, nullptr);
  gemm(PIH1, 128, PIW2h, 128, F(25), Bc * NAc, 64, 128, ACT_LRELU, nullptr, PIH2, nullptr);
  gemm(PIH2, 64, PIW3h, 64, F(27), Bc * NAc, 64, 64, ACT_NONE, PW, nullptr, nullptr);
  k_sumatoms<<<g1(Bc * HDc), 256, 0, stream>>>(PW, ZI0);
  k_dense_small<<<g1(Bc * 64), 256, 0, stream>>>(ZI0, F(28), F(29), Bc, 64, 64, ACT_LRELU, ZI1);
  k_dense_small<<<g1(Bc * 64), 256, 0, stream>>>(ZI1, F(30), F(31), Bc, 64, 64, ACT_LRELU, ZI2);
  k_dense_small<<<g1(Bc * 64), 256, 0, stream>>>(ZI2, F(32), F(33), Bc, 64, 64, ACT_NONE, dout + Bc * NCc * 3);
}
